// RNN_64235530879484
// MI455X (gfx1250) — matrix operand order repaired and checked by reading, not compiled, not run on hardware
//
#include <hip/hip_runtime.h>

typedef __attribute__((ext_vector_type(16))) __bf16 v16bf;
typedef __attribute__((ext_vector_type(8)))  __bf16 v8bf;
typedef __attribute__((ext_vector_type(8)))  float  v8f;

#define T_STEPS 1000
#define BATCH   512
#define HID     512
#define NIN     13
#define ALPHA_F 0.2f
#define SIGMA_F 0.15811388300841897f   // sqrt(2/0.2)*0.05

__device__ __forceinline__ unsigned short f32_to_bf16_bits(float x) {
    unsigned u = __builtin_bit_cast(unsigned, x);
    unsigned r = u + 0x7FFFu + ((u >> 16) & 1u);   // round-to-nearest-even
    return (unsigned short)(r >> 16);
}

// ---------------------------------------------------------------------------
// Kernel 0: re-mask diagonal of W_hh, convert to bf16, and swizzle into WMMA
// B-fragment order: frag(kc,nt) = 32 lanes x 16 bf16, lane = (k%32)/16*16 + n%16,
// element j = k%16.  One fully-coalesced 1KB block per fragment.
// ---------------------------------------------------------------------------
__global__ void prep_whh(const float* __restrict__ whh,
                         unsigned short* __restrict__ wsB) {
    int idx = blockIdx.x * blockDim.x + threadIdx.x;      // 0..512*512-1
    int k = idx >> 9;
    int n = idx & (HID - 1);
    float v = (k == n) ? 0.0f : whh[idx];
    int kc   = k >> 5, kr = k & 31;
    int half = kr >> 4, j  = kr & 15;
    int nt   = n >> 4,  nl = n & 15;
    int lanei = half * 16 + nl;
    int frag  = kc * 32 + nt;
    wsB[(size_t)frag * 512 + lanei * 16 + j] = f32_to_bf16_bits(v);
}

// ---------------------------------------------------------------------------
// Persistent scan kernel: 32 workgroups x 256 threads.  Block b owns batch
// rows [16b, 16b+16).  Each of 8 waves owns 4 N-tiles (64 columns).  State is
// carried in f32 WMMA-accumulator registers across all 1000 steps.
// ---------------------------------------------------------------------------
__global__ __launch_bounds__(256) void rnn_scan(
    const float* __restrict__ inputs,          // [T,B,I]
    const float* __restrict__ init_state,      // [B,H]
    const float* __restrict__ noise,           // [T,B,H]
    const float* __restrict__ wih,             // [I,H]
    const float* __restrict__ bias,            // [1,H]
    const unsigned short* __restrict__ wsBu,   // swizzled bf16 W_hh fragments
    float* __restrict__ out)                   // [T+1,B,H]
{
    __shared__ __align__(16) __bf16 sA[16][HID];   // relu(state) bf16, 16KB
    __shared__ float sWih[NIN][HID];               // 26KB, resident whole scan
    __shared__ float sInp[16][NIN];                // current input tile

    const __bf16* __restrict__ wsB = (const __bf16*)wsBu;

    const int tid  = threadIdx.x;
    const int lane = tid & 31;
    const int wave = tid >> 5;            // 0..7
    const int nl   = lane & 15;           // column within 16-wide tile
    const int mofs = (lane >> 4) * 8;     // C/D row offset for upper lanes
    const int rb   = blockIdx.x * 16;     // first batch row of this block

    // W_ih -> LDS (once)
    for (int i = tid; i < NIN * HID; i += 256)
        sWih[i / HID][i % HID] = wih[i];

    // out[0] = initial_state (this block's rows)
    for (int i = tid; i < 16 * HID; i += 256) {
        int r = i >> 9, c = i & (HID - 1);
        out[(size_t)(rb + r) * HID + c] = init_state[(size_t)(rb + r) * HID + c];
    }

    // load state into accumulator-layout registers; cache bias per column
    float st[4][8];
    float bcol[4];
    int   colv[4];
    #pragma unroll
    for (int q = 0; q < 4; ++q) {
        int col = wave * 64 + q * 16 + nl;
        colv[q] = col;
        bcol[q] = bias[col];
        #pragma unroll
        for (int r = 0; r < 8; ++r)
            st[q][r] = init_state[(size_t)(rb + r + mofs) * HID + col];
    }

    __syncthreads();

    for (int t = 0; t < T_STEPS; ++t) {
        // --- stage relu(state) as bf16 A-operand into LDS (row-major) ---
        #pragma unroll
        for (int q = 0; q < 4; ++q)
            #pragma unroll
            for (int r = 0; r < 8; ++r) {
                float v = st[q][r];
                v = v > 0.0f ? v : 0.0f;
                sA[r + mofs][colv[q]] = __builtin_bit_cast(__bf16, f32_to_bf16_bits(v));
            }
        // --- stage this step's input tile (16 rows x 13) ---
        if (tid < 16 * NIN) {
            int rr = tid / NIN, ii = tid % NIN;
            sInp[rr][ii] = inputs[((size_t)t * BATCH + rb + rr) * NIN + ii];
        }
        __syncthreads();

        // --- seed C with bias + fused x_proj + sigma*noise ---
        v8f acc[4];
        #pragma unroll
        for (int q = 0; q < 4; ++q) {
            float x[8];
            #pragma unroll
            for (int r = 0; r < 8; ++r) x[r] = bcol[q];
            #pragma unroll
            for (int i = 0; i < NIN; ++i) {
                float wv = sWih[i][colv[q]];
                #pragma unroll
                for (int r = 0; r < 8; ++r)
                    x[r] += sInp[r + mofs][i] * wv;
            }
            #pragma unroll
            for (int r = 0; r < 8; ++r) {
                float nz = noise[((size_t)t * BATCH + rb + r + mofs) * HID + colv[q]];
                acc[q][r] = x[r] + SIGMA_F * nz;
            }
        }

        // --- relu(state) @ W_hh : 16 k-chunks, A-frag reused across 4 tiles ---
        #pragma unroll 4
        for (int kc = 0; kc < 16; ++kc) {
            const __bf16* ap = &sA[nl][kc * 32 + mofs * 2]; // two ds_load_b128
            v8bf a0 = *(const v8bf*)ap;
            v8bf a1 = *(const v8bf*)(ap + 8);
            v16bf afrag = __builtin_shufflevector(a0, a1,
                0, 1, 2, 3, 4, 5, 6, 7, 8, 9, 10, 11, 12, 13, 14, 15);
            #pragma unroll
            for (int q = 0; q < 4; ++q) {
                int ntile = wave * 4 + q;
                const __bf16* bp = wsB + ((size_t)(kc * 32 + ntile) * 512 + lane * 16);
                v8bf b0 = *(const v8bf*)bp;                 // global_load_b128 x2
                v8bf b1 = *(const v8bf*)(bp + 8);
                v16bf bfrag = __builtin_shufflevector(b0, b1,
                    0, 1, 2, 3, 4, 5, 6, 7, 8, 9, 10, 11, 12, 13, 14, 15);
                acc[q] = __builtin_amdgcn_wmma_f32_16x16x32_bf16(
                    false, afrag, false, bfrag, (short)0, acc[q], false, false);
            }
        }

        // --- leaky integration + store state[t+1] ---
        float* orow = out + (size_t)(t + 1) * BATCH * HID;
        #pragma unroll
        for (int q = 0; q < 4; ++q)
            #pragma unroll
            for (int r = 0; r < 8; ++r) {
                float s = (1.0f - ALPHA_F) * st[q][r] + ALPHA_F * acc[q][r];
                st[q][r] = s;
                orow[(size_t)(rb + r + mofs) * HID + colv[q]] = s;
            }
        __syncthreads();   // protect sA/sInp before next iteration rewrites
    }
}

extern "C" void kernel_launch(void* const* d_in, const int* in_sizes, int n_in,
                              void* d_out, int out_size, void* d_ws, size_t ws_size,
                              hipStream_t stream) {
    const float* inputs = (const float*)d_in[0];
    const float* init   = (const float*)d_in[1];
    const float* noise  = (const float*)d_in[2];
    const float* wih    = (const float*)d_in[3];
    const float* whh    = (const float*)d_in[4];
    const float* bias   = (const float*)d_in[5];
    float* out = (float*)d_out;
    unsigned short* wsB = (unsigned short*)d_ws;   // 512*512*2 = 512 KB

    prep_whh<<<(HID * HID) / 256, 256, 0, stream>>>(whh, wsB);
    rnn_scan<<<BATCH / 16, 256, 0, stream>>>(inputs, init, noise, wih, bias, wsB, out);
}
